// MResConvFace_12790412607758
// MI455X (gfx1250) — matrix operand order repaired and checked by reading, not compiled, not run on hardware
//
#include <hip/hip_runtime.h>

// ---------------------------------------------------------------------------
// MResConvFace on gfx1250 (MI455X).
// Core math per conv = skinny GEMM [B*F, 96] x [96, 32] -> lowered to
// v_wmma_f32_16x16x32_f16 (M=16 faces, N=16 out-ch tile x2, K=96 in 3 steps).
// Gathers hit L2 (82MB x-tensor < 192MB L2). fp16 A/B, fp32 accumulate.
// ---------------------------------------------------------------------------

typedef __attribute__((ext_vector_type(16))) _Float16 v16h;
typedef __attribute__((ext_vector_type(8)))  float    v8f;

#define C_CH        32           // C_in == C_out == 32
#define KRED        96           // C * 3 symmetric features
#define FEAT_STRIDE 104          // padded LDS row stride in halfs (16B multiple)
#define WAVES       4            // waves per block (wave32)
#define TILE_M      16           // faces per wave
#define BN_EPS      1e-5f

// Load a v16h A/B fragment as two 16-byte chunks (halfs 0..7 | 8..15).
__device__ __forceinline__ v16h ld_frag2(const _Float16* p0, const _Float16* p1) {
  union { v16h v; struct { float4 a; float4 b; } q; } u;
  u.q.a = *(const float4*)p0;
  u.q.b = *(const float4*)p1;
  return u.v;
}

// ---------------------------------------------------------------------------
// Prep: transpose weights [O,C,K] -> f16 rows Bp[o][c*3+k]; zero BN stat accums.
// ---------------------------------------------------------------------------
__global__ void prep_kernel(const float* __restrict__ w0, const float* __restrict__ w1,
                            _Float16* __restrict__ Bp0, _Float16* __restrict__ Bp1,
                            float* __restrict__ stats) {
  int i = blockIdx.x * blockDim.x + threadIdx.x;
  if (i < 64) stats[i] = 0.0f;                       // sums[32], sumsq[32]
  if (i < C_CH * KRED) {
    int o = i / KRED, kred = i % KRED;
    int c = kred / 3, k = kred % 3;
    Bp0[i] = (_Float16)w0[(o * C_CH + c) * 3 + k];
    Bp1[i] = (_Float16)w1[(o * C_CH + c) * 3 + k];
  }
}

// ---------------------------------------------------------------------------
// Finalize BN: fold mean/var of relu(y0) into per-channel affine a,b.
// ---------------------------------------------------------------------------
__global__ void finalize_kernel(const float* __restrict__ sums, const float* __restrict__ sumsq,
                                const float* __restrict__ gamma, const float* __restrict__ beta,
                                float* __restrict__ ascale, float* __restrict__ bshift,
                                float invN) {
  int o = threadIdx.x;
  if (o < C_CH) {
    float mean = sums[o] * invN;
    float var  = sumsq[o] * invN - mean * mean;
    float a    = gamma[o] * rsqrtf(var + BN_EPS);
    ascale[o]  = a;
    bshift[o]  = beta[o] - mean * a;
  }
}

// ---------------------------------------------------------------------------
// Fused mesh conv. MODE 0: y0 = conv0(x), accumulate relu stats.
//                  MODE 1: out = relu(conv1(a*relu(y0)+b) + y0).
// Each wave: 16 faces x 32 out channels, K=96 via 3x v_wmma_f32_16x16x32_f16
// per N-tile (6 WMMAs). Features staged in LDS in the A-matrix 16-bit layout.
// ---------------------------------------------------------------------------
template <int MODE>
__global__ void __launch_bounds__(WAVES * 32)
mesh_conv_kernel(const float* __restrict__ X,          // [B,32,F] (x or y0)
                 const long long* __restrict__ gemm,   // [B,F,3] int64
                 const _Float16* __restrict__ Bp,      // [32][96] f16 weights
                 float* __restrict__ Y,                // [B,32,F] output
                 const float* __restrict__ ascale,     // [32] (MODE1)
                 const float* __restrict__ bshift,     // [32] (MODE1)
                 float* __restrict__ sums,             // [32] (MODE0)
                 float* __restrict__ sumsq,            // [32] (MODE0)
                 int F) {
  __shared__ __align__(16) _Float16 feats[WAVES][TILE_M * FEAT_STRIDE];
  __shared__ __align__(16) float    outT[WAVES][C_CH * TILE_M];

  const int tid  = threadIdx.x;
  const int wave = tid >> 5;
  const int lane = tid & 31;
  const int m    = lane & 15;    // face row within tile / A-matrix M
  const int h    = lane >> 4;    // lane half -> K-group select (A/B layouts)

  const int b  = blockIdx.y;
  const int f0 = (blockIdx.x * WAVES + wave) * TILE_M;
  int f = f0 + m;
  if (f >= F) f = F - 1;         // clamp loads; stores are guarded below

  // ---- B-matrix fragments from pre-transposed f16 weights (registers) ----
  // Tile t covers out channels o = 16t + n, n = lane&15.
  // lane<16 holds K = s*32 + [0..15]; lane>=16 holds K = s*32 + [16..31].
  v16h bf[3][2];
#pragma unroll
  for (int t = 0; t < 2; ++t) {
    const _Float16* brow = Bp + (size_t)(t * 16 + m) * KRED;
#pragma unroll
    for (int s = 0; s < 3; ++s) {
      const _Float16* p = brow + s * 32 + h * 16;
      bf[s][t] = ld_frag2(p, p + 8);
    }
  }

  // ---- gather neighbors + build symmetric features into LDS (f16) ----
  const long long* g = gemm + ((size_t)b * F + f) * 3;
  const int j0 = (int)g[0], j1 = (int)g[1], j2 = (int)g[2];

  const size_t xbase = (size_t)b * C_CH * F;
  _Float16* frow = feats[wave] + m * FEAT_STRIDE;
  const int c0 = h * 16;                     // this lane covers 16 channels
#pragma unroll 4
  for (int i = 0; i < 16; ++i) {
    const int c = c0 + i;
    const float* xc = X + xbase + (size_t)c * F;
    float ctr = xc[f];
    float a0 = xc[j0], a1 = xc[j1], a2 = xc[j2];
    if (MODE) {                              // z = a*relu(y0) + b (fused BN)
      const float as = ascale[c], bs = bshift[c];
      ctr = fmaxf(ctr, 0.f) * as + bs;
      a0  = fmaxf(a0, 0.f) * as + bs;
      a1  = fmaxf(a1, 0.f) * as + bs;
      a2  = fmaxf(a2, 0.f) * as + bs;
    }
    const float s1 = a0 + a1 + a2;
    const float s2 = fabsf(a0 - a1) + fabsf(a1 - a2) + fabsf(a2 - a0);
    frow[c * 3 + 0] = (_Float16)ctr;
    frow[c * 3 + 1] = (_Float16)s1;
    frow[c * 3 + 2] = (_Float16)s2;
  }
  __syncthreads();

  // ---- WMMA: A from LDS (contiguous K order used here, same as B: lane<16 K{0..15}, else
  //            K{16..31} per 32-K step), 3 K-steps x 2 N-tiles ----
  v8f acc0 = {};
  v8f acc1 = {};
  const _Float16* arow = feats[wave] + m * FEAT_STRIDE;
#pragma unroll
  for (int s = 0; s < 3; ++s) {
    v16h af = ld_frag2(arow + s * 32 + h * 16,
                       arow + s * 32 + h * 16 + 8);
    acc0 = __builtin_amdgcn_wmma_f32_16x16x32_f16(false, af, false, bf[s][0],
                                                  (short)0, acc0, false, false);
    acc1 = __builtin_amdgcn_wmma_f32_16x16x32_f16(false, af, false, bf[s][1],
                                                  (short)0, acc1, false, false);
  }

  // ---- D layout (lane<16: M=v, N=lane; lane>=16: M=v+8) -> LDS transpose ----
  {
    float* ot = outT[wave];
    const int mb = h * 8;
#pragma unroll
    for (int v = 0; v < 8; ++v) {
      ot[(m     ) * TILE_M + (mb + v)] = acc0[v];
      ot[(m + 16) * TILE_M + (mb + v)] = acc1[v];
    }
  }
  __syncthreads();

  // ---- coalesced writeback: lane -> out channel o, 16 contiguous faces ----
  {
    const int o = lane;
    const float* ot   = outT[wave] + o * TILE_M;
    float*       yrow = Y + xbase + (size_t)o * F + f0;
    const float* rrow = X + xbase + (size_t)o * F + f0;   // residual (MODE1: X==y0)
    float s1 = 0.f, s2 = 0.f;
#pragma unroll
    for (int j = 0; j < TILE_M; ++j) {
      if (f0 + j >= F) break;
      float v = ot[j];
      if (MODE) {
        v = fmaxf(v + rrow[j], 0.f);        // residual + relu
      } else {
        float r = fmaxf(v, 0.f);            // stats over relu(y0)
        s1 += r;
        s2 += r * r;
      }
      yrow[j] = v;                          // MODE0 stores raw y0 (pre-relu)
    }
    if (!MODE && f0 < F) {
      atomicAdd(&sums[o], s1);
      atomicAdd(&sumsq[o], s2);
    }
  }
}

// ---------------------------------------------------------------------------
// Launch: prep -> conv0(+stats) -> BN finalize -> conv1(+residual relu)
// ---------------------------------------------------------------------------
extern "C" void kernel_launch(void* const* d_in, const int* in_sizes, int n_in,
                              void* d_out, int out_size, void* d_ws, size_t ws_size,
                              hipStream_t stream) {
  const float*     x     = (const float*)d_in[0];      // [B,32,F]
  const long long* gemm  = (const long long*)d_in[1];  // [B,F,3] int64
  const float*     w0    = (const float*)d_in[2];      // [32,32,3]
  const float*     w1    = (const float*)d_in[3];      // [32,32,3]
  const float*     gamma = (const float*)d_in[4];      // [32]
  const float*     beta  = (const float*)d_in[5];      // [32]

  const int B = 16;
  const int F = in_sizes[0] / (B * C_CH);              // 40000

  // Workspace: y0 [B,32,F] f32, then f16 weight tables, then stat/affine buf.
  float*     y0   = (float*)d_ws;
  size_t     y0n  = (size_t)B * C_CH * F;
  _Float16*  Bp0  = (_Float16*)(y0 + y0n);
  _Float16*  Bp1  = Bp0 + C_CH * KRED;
  float*     stats = (float*)(Bp1 + C_CH * KRED);
  float* sums  = stats;        // [32]
  float* sumsq = stats + 32;   // [32]
  float* asc   = stats + 64;   // [32]
  float* bsh   = stats + 96;   // [32]

  prep_kernel<<<dim3((C_CH * KRED + 255) / 256), 256, 0, stream>>>(
      w0, w1, Bp0, Bp1, stats);

  dim3 grid((F + WAVES * TILE_M - 1) / (WAVES * TILE_M), B);
  dim3 block(WAVES * 32);

  mesh_conv_kernel<0><<<grid, block, 0, stream>>>(
      x, gemm, Bp0, y0, nullptr, nullptr, sums, sumsq, F);

  finalize_kernel<<<1, 32, 0, stream>>>(
      sums, sumsq, gamma, beta, asc, bsh, 1.0f / ((float)B * (float)F));

  mesh_conv_kernel<1><<<grid, block, 0, stream>>>(
      y0, gemm, Bp1, (float*)d_out, asc, bsh, nullptr, nullptr, F);
}
